// FPN2MLPFeatureExtractor_24979529794325
// MI455X (gfx1250) — matrix operand order repaired and checked by reading, not compiled, not run on hardware
//
#include <hip/hip_runtime.h>

// ---------------------------------------------------------------------------
// MI455X (gfx1250): ROIAlign3D -> conv-GEMM -> BN/ReLU -> fc6 GEMM (split-K)
// -> fc7 GEMM.  GEMMs: v_wmma_f32_16x16x32_bf16 with TDM (tensor_load_to_lds)
// double-buffered A-tile prefetch + s_wait_tensorcnt.
// ---------------------------------------------------------------------------

typedef __attribute__((ext_vector_type(16))) __bf16       bf16x16;
typedef __attribute__((ext_vector_type(8)))  float        v8f;
typedef __attribute__((ext_vector_type(4)))  unsigned int u32x4;
typedef __attribute__((ext_vector_type(2)))  unsigned int u32x2;
typedef __attribute__((ext_vector_type(4)))  float        f32x4;
typedef __attribute__((ext_vector_type(4)))  int          i32x4;
typedef __attribute__((ext_vector_type(8)))  int          i32x8;

union FragBF { bf16x16 v; u32x4 q[2]; };

#define C_DIM 128
#define XD 80
#define YD 80
#define ZD 20
#define RESV 7
#define NROI 1024
#define REP 896
#define MCONV (NROI * RESV * RESV)          // 50176
#define KCONV (C_DIM * RESV)                // 896
#define KFC6  (C_DIM * RESV * RESV * RESV)  // 43904
#define BM 128
#define BN 128
#define BK 32
#define LDSW 40  // padded bf16 row stride (64B data + 16B pad) per tile row

// ---------------------------------------------------------------------------
// TDM: issue tensor_load_to_lds for one 128x32 bf16 tile (row stride = K
// elements) into LDS at lds_off, with hardware padding of 4 DWORDs after
// every 16 DWORDs -> reproduces the LDSW=40 padded layout.
// D# per CDNA5 ISA  (2-group form, <=2D tensor).
// ---------------------------------------------------------------------------
__device__ __forceinline__ void tdm_load_tile_a(const __bf16* gptr, unsigned lds_off, int K)
{
    const unsigned long long ga = (unsigned long long)(size_t)gptr;
    i32x4 g0;
    g0[0] = 1;                                                  // count=1 (valid user D#)
    g0[1] = (int)lds_off;                                       // lds_addr
    g0[2] = (int)(unsigned)(ga & 0xffffffffull);                // global_addr[31:0]
    g0[3] = (int)((unsigned)((ga >> 32) & 0x01ffffffull) | (2u << 30)); // addr[56:32] | type=2

    const unsigned td0 = 0x40000000u;   // tensor_dim0 (no OOB for interior tiles)
    const unsigned td1 = 0x40000000u;   // tensor_dim1
    i32x8 g1;
    g1[0] = (1 << 16)                   // data_size = 2 bytes
          | (1 << 20)                   // pad_enable
          | (3 << 22)                   // pad_interval: 16 DWORDs (=64B row)
          | (3 << 25);                  // pad_amount: 4 DWORDs (=16B)
    g1[1] = (int)((td0 & 0xffffu) << 16);                       // tensor_dim0[15:0]
    g1[2] = (int)(((td0 >> 16) & 0xffffu) | ((td1 & 0xffffu) << 16));
    g1[3] = (int)(((td1 >> 16) & 0xffffu) | (32u << 16));       // tile_dim0 = 32
    g1[4] = 128;                                                // tile_dim1 = 128, tile_dim2 = 0
    g1[5] = K;                                                  // tensor_dim0_stride[31:0]
    g1[6] = 0;                                                  // stride hi / dim1_stride
    g1[7] = 0;
    asm volatile("tensor_load_to_lds %0, %1" :: "s"(g0), "s"(g1) : "memory");
}

// ---------------------------------------------------------------------------
// Kernel 1: 3D ROIAlign (sampling_ratio=1), writes bf16 pooled in GEMM-A
// layout: pooled[(n*49 + x*7 + y) * 896 + c*7 + z]
// ---------------------------------------------------------------------------
__global__ void __launch_bounds__(128)
roi_align3d_kernel(const float* __restrict__ feat, const float* __restrict__ boxes,
                   const int* __restrict__ batch_idx, __bf16* __restrict__ pooled)
{
    const int blk = blockIdx.x;            // n*343 + p
    const int n = blk / 343;
    const int p = blk - n * 343;
    const int xb = p / 49;
    const int yb = (p / 7) % 7;
    const int zb = p % 7;
    const int c = threadIdx.x;             // 0..127

    const float s0 = boxes[n * 6 + 0] * 5.0f;   // VOXEL_SCALE * POOLER_SCALE
    const float s1 = boxes[n * 6 + 1] * 5.0f;
    const float s2 = boxes[n * 6 + 2] * 5.0f;
    const float e0 = boxes[n * 6 + 3] * 5.0f;
    const float e1 = boxes[n * 6 + 4] * 5.0f;
    const float e2 = boxes[n * 6 + 5] * 5.0f;
    const float bsx = fmaxf(e0 - s0, 1.0f) * (1.0f / RESV);
    const float bsy = fmaxf(e1 - s1, 1.0f) * (1.0f / RESV);
    const float bsz = fmaxf(e2 - s2, 1.0f) * (1.0f / RESV);
    const float px = s0 + ((float)xb + 0.5f) * bsx;
    const float py = s1 + ((float)yb + 0.5f) * bsy;
    const float pz = s2 + ((float)zb + 0.5f) * bsz;

    const float fx = floorf(px), fy = floorf(py), fz = floorf(pz);
    const float lx = px - fx, ly = py - fy, lz = pz - fz;
    const int ix0 = (int)fx, iy0 = (int)fy, iz0 = (int)fz;
    int xs[2], ys[2], zs[2];
    xs[0] = ix0 < 0 ? 0 : (ix0 > XD - 1 ? XD - 1 : ix0);
    xs[1] = ix0 + 1 < 0 ? 0 : (ix0 + 1 > XD - 1 ? XD - 1 : ix0 + 1);
    ys[0] = iy0 < 0 ? 0 : (iy0 > YD - 1 ? YD - 1 : iy0);
    ys[1] = iy0 + 1 < 0 ? 0 : (iy0 + 1 > YD - 1 ? YD - 1 : iy0 + 1);
    zs[0] = iz0 < 0 ? 0 : (iz0 > ZD - 1 ? ZD - 1 : iz0);
    zs[1] = iz0 + 1 < 0 ? 0 : (iz0 + 1 > ZD - 1 ? ZD - 1 : iz0 + 1);
    const float wx[2] = {1.0f - lx, lx};
    const float wy[2] = {1.0f - ly, ly};
    const float wz[2] = {1.0f - lz, lz};

    const float* fb = feat + (size_t)batch_idx[n] * XD * YD * ZD * C_DIM;
    float acc = 0.0f;
#pragma unroll
    for (int dx = 0; dx < 2; ++dx)
#pragma unroll
        for (int dy = 0; dy < 2; ++dy)
#pragma unroll
            for (int dz = 0; dz < 2; ++dz) {
                const float w = wx[dx] * wy[dy] * wz[dz];
                const size_t lin = (((size_t)xs[dx] * YD + ys[dy]) * ZD + zs[dz]) * C_DIM + c;
                acc += w * fb[lin];
            }

    const size_t row = (size_t)n * 49 + xb * 7 + yb;
    pooled[row * KCONV + c * 7 + zb] = (__bf16)acc;
}

// ---------------------------------------------------------------------------
// Kernel 2: tiled bf16 WMMA GEMM with TDM double-buffered A-tile prefetch.
// A: bf16 [M][K] (lda=K).
// Bsrc f32:  bLayout 0 -> B[k*ldb + n]   (fc6_w, fc7_w)
//            bLayout 1 -> B[n*ldb + k]   (conv_w as [rep][C*7])
// epi 0: +bias, store bf16 C, per-column sum/sumsq atomics (BN stats)
// epi 2: +bias, ReLU, store f32 C
// epi 3: store f32 into split-K slice blockIdx.z (no bias) -- deterministic
// ---------------------------------------------------------------------------
__global__ void __launch_bounds__(256)
wmma_gemm_kernel(const __bf16* __restrict__ A, const float* __restrict__ Bsrc,
                 const float* __restrict__ bias, void* __restrict__ Cout,
                 float* __restrict__ stats,
                 int M, int N, int K, int ldb, int bLayout, int epi, int kChunk)
{
    __shared__ __align__(16) __bf16 Asm[2][BM * LDSW];
    __shared__ __align__(16) __bf16 Bsm[2][BN * LDSW];

    const int tid   = threadIdx.x;
    const int lane  = tid & 31;
    const int wave  = tid >> 5;               // 0..7
    const int waveM = (wave >> 2) * 64;       // 0 or 64
    const int waveN = (wave & 3) * 32;        // 0,32,64,96
    const int lhalf = lane >> 4;              // 0/1
    const int lmod  = lane & 15;
    const int rowBase = blockIdx.x * BM;
    const int colBase = blockIdx.y * BN;
    const int kBegin  = blockIdx.z * kChunk;
    const int kLim    = kBegin + kChunk;
    const int kEnd    = kLim < K ? kLim : K;

    v8f acc[4][2];
#pragma unroll
    for (int mi = 0; mi < 4; ++mi)
#pragma unroll
        for (int ni = 0; ni < 2; ++ni) {
            v8f z = {0.f, 0.f, 0.f, 0.f, 0.f, 0.f, 0.f, 0.f};
            acc[mi][ni] = z;
        }

    const __bf16* Atile = A + (size_t)rowBase * K;

    // Prologue: TDM prefetch of the first A tile (wave 0; EXEC-independent op)
    if (wave == 0)
        tdm_load_tile_a(Atile + kBegin, (unsigned)(size_t)&Asm[0][0], K);

    int buf = 0;
    for (int k0 = kBegin; k0 < kEnd; k0 += BK) {
        // ---- B tile -> LDS[buf], transposed to Bt[n][k], f32 -> bf16 ----
        if (bLayout == 0) {
            const int kk = tid >> 5;          // 0..7
            const int nq = lane << 2;         // 0..124
#pragma unroll
            for (int i = 0; i < 4; ++i) {
                const int k = kk + i * 8;
                const f32x4 d = *reinterpret_cast<const f32x4*>(
                    Bsrc + (size_t)(k0 + k) * ldb + (colBase + nq));
#pragma unroll
                for (int j = 0; j < 4; ++j)
                    Bsm[buf][(nq + j) * LDSW + k] = (__bf16)d[j];
            }
        } else {
            const int n0 = tid >> 3;          // 0..31
            const int kq = (tid & 7) << 2;    // 0..28
#pragma unroll
            for (int i = 0; i < 4; ++i) {
                const int nn = n0 + i * 32;
                const f32x4 d = *reinterpret_cast<const f32x4*>(
                    Bsrc + (size_t)(colBase + nn) * ldb + (k0 + kq));
#pragma unroll
                for (int j = 0; j < 4; ++j)
                    Bsm[buf][nn * LDSW + kq + j] = (__bf16)d[j];
            }
        }

        // A tile[buf] complete (TDM) + B tile[buf] visible to all waves
        __builtin_amdgcn_s_wait_tensorcnt(0);
        __syncthreads();

        // Prefetch next A tile into the other buffer while we compute.
        // (Safe: buf^1's last readers finished before the barrier above.)
        if (wave == 0 && (k0 + BK) < kEnd)
            tdm_load_tile_a(Atile + (k0 + BK), (unsigned)(size_t)&Asm[buf ^ 1][0], K);

        // ---- fragments per ISA 7.12.2 bf16 layouts; two b128 LDS reads each
        FragBF afr[4], bfr[2];
#pragma unroll
        for (int mi = 0; mi < 4; ++mi) {
            const __bf16* p = &Asm[buf][(waveM + mi * 16 + lmod) * LDSW + lhalf * 8];
            afr[mi].q[0] = *reinterpret_cast<const u32x4*>(p);
            afr[mi].q[1] = *reinterpret_cast<const u32x4*>(p + 16);
        }
#pragma unroll
        for (int ni = 0; ni < 2; ++ni) {
            const __bf16* p = &Bsm[buf][(waveN + ni * 16 + lmod) * LDSW + lhalf * 8];
            bfr[ni].q[0] = *reinterpret_cast<const u32x4*>(p);
            bfr[ni].q[1] = *reinterpret_cast<const u32x4*>(p + 16);
        }
#pragma unroll
        for (int mi = 0; mi < 4; ++mi)
#pragma unroll
            for (int ni = 0; ni < 2; ++ni)
                acc[mi][ni] = __builtin_amdgcn_wmma_f32_16x16x32_bf16(
                    false, afr[mi].v, false, bfr[ni].v,
                    (short)0, acc[mi][ni], false, false);

        buf ^= 1;
    }

    // ---- epilogues ----
    if (epi == 0) {
        __bf16* Cb = (__bf16*)Cout;
#pragma unroll
        for (int ni = 0; ni < 2; ++ni) {
            const int col = colBase + waveN + ni * 16 + lmod;
            const float bv = bias[col];
            float s = 0.f, s2 = 0.f;
#pragma unroll
            for (int mi = 0; mi < 4; ++mi) {
                const int rb = rowBase + waveM + mi * 16 + lhalf * 8;
#pragma unroll
                for (int r = 0; r < 8; ++r) {
                    const float v = acc[mi][ni][r] + bv;
                    Cb[(size_t)(rb + r) * N + col] = (__bf16)v;
                    s += v; s2 += v * v;
                }
            }
            s  += __shfl_xor(s, 16, 32);
            s2 += __shfl_xor(s2, 16, 32);
            if (lhalf == 0) {
                atomicAdd(&stats[col], s);
                atomicAdd(&stats[N + col], s2);
            }
        }
    } else if (epi == 2) {
        float* Cf = (float*)Cout;
#pragma unroll
        for (int ni = 0; ni < 2; ++ni) {
            const int col = colBase + waveN + ni * 16 + lmod;
            const float bv = bias[col];
#pragma unroll
            for (int mi = 0; mi < 4; ++mi) {
                const int rb = rowBase + waveM + mi * 16 + lhalf * 8;
#pragma unroll
                for (int r = 0; r < 8; ++r)
                    Cf[(size_t)(rb + r) * N + col] = fmaxf(acc[mi][ni][r] + bv, 0.f);
            }
        }
    } else { // epi == 3 : deterministic split-K slice
        float* Cf = (float*)Cout + (size_t)blockIdx.z * M * N;
#pragma unroll
        for (int ni = 0; ni < 2; ++ni) {
            const int col = colBase + waveN + ni * 16 + lmod;
#pragma unroll
            for (int mi = 0; mi < 4; ++mi) {
                const int rb = rowBase + waveM + mi * 16 + lhalf * 8;
#pragma unroll
                for (int r = 0; r < 8; ++r)
                    Cf[(size_t)(rb + r) * N + col] = acc[mi][ni][r];
            }
        }
    }
}

// ---------------------------------------------------------------------------
// BN finalize: per-channel scale/shift from sum/sumsq
// ---------------------------------------------------------------------------
__global__ void __launch_bounds__(896)
bn_stats_kernel(const float* __restrict__ stats, const float* __restrict__ gamma,
                const float* __restrict__ beta, float* __restrict__ ss)
{
    const int r = threadIdx.x;
    if (r < REP) {
        const float inv = 1.0f / (float)MCONV;
        const float mu  = stats[r] * inv;
        const float var = stats[REP + r] * inv - mu * mu;
        const float sc  = gamma[r] * rsqrtf(var + 1e-5f);
        ss[r]       = sc;
        ss[REP + r] = beta[r] - mu * sc;
    }
}

// ---------------------------------------------------------------------------
// BN apply + ReLU + remap y[(n*49+xy)][r] -> x2[n][r*49+xy]  (bf16)
// ---------------------------------------------------------------------------
__global__ void __launch_bounds__(256)
bn_remap_kernel(const __bf16* __restrict__ Y, const float* __restrict__ ss,
                __bf16* __restrict__ X2)
{
    const size_t t = (size_t)blockIdx.x * 256 + threadIdx.x;  // < 50176*896
    const int row = (int)(t / REP);            // n*49 + xy
    const int r   = (int)(t - (size_t)row * REP);
    const int n   = row / 49;
    const int xy  = row - n * 49;
    float v = (float)Y[t] * ss[r] + ss[REP + r];
    v = fmaxf(v, 0.f);
    X2[(size_t)n * KFC6 + r * 49 + xy] = (__bf16)v;
}

// ---------------------------------------------------------------------------
// fc6 split-K reduce + bias + ReLU -> bf16 x3
// ---------------------------------------------------------------------------
__global__ void __launch_bounds__(256)
fc6_reduce_kernel(const float* __restrict__ part, const float* __restrict__ bias,
                  __bf16* __restrict__ x3)
{
    const int t = blockIdx.x * 256 + threadIdx.x;  // < 1024*896
    const int c = t % REP;
    float s = bias[c];
#pragma unroll
    for (int z = 0; z < 7; ++z)
        s += part[(size_t)z * NROI * REP + t];
    x3[t] = (__bf16)fmaxf(s, 0.f);
}

__global__ void __launch_bounds__(256)
zero_f32_kernel(float* __restrict__ p, int n)
{
    const int i = blockIdx.x * 256 + threadIdx.x;
    if (i < n) p[i] = 0.f;
}

// ---------------------------------------------------------------------------
extern "C" void kernel_launch(void* const* d_in, const int* in_sizes, int n_in,
                              void* d_out, int out_size, void* d_ws, size_t ws_size,
                              hipStream_t stream) {
    const float* features  = (const float*)d_in[0];
    const float* boxes     = (const float*)d_in[1];
    const float* conv_w    = (const float*)d_in[2];
    const float* conv_b    = (const float*)d_in[3];
    const float* gamma     = (const float*)d_in[4];
    const float* beta      = (const float*)d_in[5];
    const float* fc6_w     = (const float*)d_in[6];
    const float* fc6_b     = (const float*)d_in[7];
    const float* fc7_w     = (const float*)d_in[8];
    const float* fc7_b     = (const float*)d_in[9];
    const int*   batch_idx = (const int*)d_in[10];
    float* out = (float*)d_out;

    char* ws = (char*)d_ws;
    const size_t P = (size_t)MCONV * KCONV * 2;          // 89,915,392 B
    __bf16* pooled = (__bf16*)ws;                        // A for conv; later x2
    __bf16* ybf    = (__bf16*)(ws + P);                  // conv output bf16
    float*  part   = (float*)(ws + 2 * P);               // 7 split-K slices
    const size_t partB = (size_t)7 * NROI * REP * 4;     // 25,690,112 B
    __bf16* x3     = (__bf16*)(ws + 2 * P + partB);
    float*  stats  = (float*)(ws + 2 * P + partB + (size_t)NROI * REP * 2);
    float*  ss     = stats + 2 * REP;                    // scale, shift

    // 1) init BN stats accumulators
    zero_f32_kernel<<<(2 * REP + 255) / 256, 256, 0, stream>>>(stats, 2 * REP);

    // 2) 3D ROIAlign -> pooled bf16 in GEMM-A layout
    roi_align3d_kernel<<<NROI * 343, 128, 0, stream>>>(features, boxes, batch_idx, pooled);

    // 3) conv einsum as GEMM (M=50176,K=896,N=896) + bias + BN stats
    wmma_gemm_kernel<<<dim3(MCONV / BM, REP / BN, 1), 256, 0, stream>>>(
        pooled, conv_w, conv_b, (void*)ybf, stats,
        MCONV, REP, KCONV, KCONV, /*bLayout=*/1, /*epi=*/0, /*kChunk=*/KCONV);

    // 4) BN scale/shift
    bn_stats_kernel<<<1, 896, 0, stream>>>(stats, gamma, beta, ss);

    // 5) BN apply + ReLU + remap to x2 (reuses pooled buffer)
    bn_remap_kernel<<<(MCONV * REP) / 256, 256, 0, stream>>>(ybf, ss, pooled);

    // 6) fc6 GEMM (M=1024,K=43904,N=896), split-K=7 for occupancy
    wmma_gemm_kernel<<<dim3(NROI / BM, REP / BN, 7), 256, 0, stream>>>(
        pooled, fc6_w, nullptr, (void*)part, nullptr,
        NROI, REP, KFC6, REP, /*bLayout=*/0, /*epi=*/3, /*kChunk=*/KFC6 / 7);

    // 7) reduce slices + bias + ReLU -> x3 bf16
    fc6_reduce_kernel<<<(NROI * REP) / 256, 256, 0, stream>>>(part, fc6_b, x3);

    // 8) fc7 GEMM (M=1024,K=896,N=896) + bias + ReLU -> f32 output
    wmma_gemm_kernel<<<dim3(NROI / BM, REP / BN, 1), 256, 0, stream>>>(
        x3, fc7_w, fc7_b, (void*)out, nullptr,
        NROI, REP, KCONV, REP, /*bLayout=*/0, /*epi=*/2, /*kChunk=*/KCONV);
}
